// DCNv2Block_DL_80101140070678
// MI455X (gfx1250) — matrix operand order repaired and checked by reading, not compiled, not run on hardware
//
#include <hip/hip_runtime.h>
#include <math.h>

// ---------------------------------------------------------------------------
// DCNv2 block for MI455X (gfx1250, wave32, WMMA + Tensor Data Mover).
// Main contraction: M=B*H*W=16384, N=C2=256, K=C1*K2=2304 via
// v_wmma_f32_16x16x32_f16. A-tiles (deformable bilinear samples) built in LDS
// on the fly; B-tiles staged by TDM (tensor_load_to_lds) double-buffered.
// ---------------------------------------------------------------------------

typedef _Float16 half16 __attribute__((ext_vector_type(16)));
typedef _Float16 half8  __attribute__((ext_vector_type(8)));
typedef _Float16 half4  __attribute__((ext_vector_type(4)));
typedef float    float8 __attribute__((ext_vector_type(8)));
typedef unsigned int u32x4 __attribute__((ext_vector_type(4)));
typedef int      i32x4 __attribute__((ext_vector_type(4)));
typedef int      i32x8 __attribute__((ext_vector_type(8)));

#define Bsz   4
#define C1    256
#define C2    256
#define Hd    64
#define Wd    64
#define HW    4096           // 64*64
#define K2n   9
#define Ktot  2304           // C1*K2n, ordered kk = k2*256 + c
#define NCHUNK 72            // Ktot / 32

#if defined(__has_builtin)
#  if __has_builtin(__builtin_amdgcn_tensor_load_to_lds)
#    define HAVE_TDM 1
#  else
#    define HAVE_TDM 0
#  endif
#else
#  define HAVE_TDM 0
#endif

// Per-(b,hw,tap) bilinear descriptor: 4 clamped corner indices + 4 weights
// (weights already fold validity masking and the sigmoid modulation mask).
struct __align__(16) Desc {
    int   i00, i01, i10, i11;
    float w00, w01, w10, w11;
};

// ---------------------------------------------------------------------------
// K0: repack w_dcn (C2,C1,3,3) f32 -> wBT[d][k2*256 + c] f16 (N-major rows).
// ---------------------------------------------------------------------------
__global__ __launch_bounds__(256) void k_pack_w(const float* __restrict__ w_dcn,
                                                _Float16* __restrict__ wBT) {
    int idx = blockIdx.x * 256 + threadIdx.x;           // over C2*Ktot
    if (idx >= C2 * Ktot) return;
    int d  = idx / Ktot;
    int kk = idx - d * Ktot;
    int k2 = kk >> 8;
    int c  = kk & 255;
    wBT[(size_t)d * Ktot + kk] = (_Float16)w_dcn[((size_t)d * C1 + c) * 9 + k2];
}

// ---------------------------------------------------------------------------
// K1: dilated 3x3 offset conv (256 -> 27) fused with descriptor generation.
// ---------------------------------------------------------------------------
__global__ __launch_bounds__(64) void k_offconv(const float* __restrict__ x,
                                                const float* __restrict__ w_off,
                                                const float* __restrict__ b_off,
                                                Desc* __restrict__ desc) {
    const int bh = blockIdx.x;
    const int b  = bh >> 6;
    const int h  = bh & 63;
    const int w  = threadIdx.x;

    __shared__ float ws[32 * 243];

    float acc[27];
#pragma unroll
    for (int o = 0; o < 27; ++o) acc[o] = 0.f;

    const float* xb = x + (size_t)b * C1 * HW;

    for (int c0 = 0; c0 < C1; c0 += 32) {
        for (int i = threadIdx.x; i < 32 * 243; i += 64) {
            int cl  = i / 243;
            int rem = i - cl * 243;
            int o   = rem / 9;
            int t   = rem - o * 9;
            ws[i] = w_off[(size_t)o * (C1 * 9) + (size_t)(c0 + cl) * 9 + t];
        }
        __syncthreads();

        for (int cl = 0; cl < 32; ++cl) {
            const float* xp = xb + (size_t)(c0 + cl) * HW;
            float xv[9];
#pragma unroll
            for (int t = 0; t < 9; ++t) {               // dilation 2, pad 2
                int iy = h + (t / 3) * 2 - 2;
                int ix = w + (t % 3) * 2 - 2;
                xv[t] = (iy >= 0 && iy < Hd && ix >= 0 && ix < Wd)
                          ? xp[iy * Wd + ix] : 0.f;
            }
            const float* wsl = ws + cl * 243;
#pragma unroll
            for (int o = 0; o < 27; ++o) {
                float s = 0.f;
#pragma unroll
                for (int t = 0; t < 9; ++t) s += xv[t] * wsl[o * 9 + t];
                acc[o] += s;
            }
        }
        __syncthreads();
    }

    const size_t dbase = ((size_t)b * HW + h * Wd + w) * K2n;
#pragma unroll
    for (int k2 = 0; k2 < K2n; ++k2) {
        float dy = acc[2 * k2]     + b_off[2 * k2];
        float dx = acc[2 * k2 + 1] + b_off[2 * k2 + 1];
        dy = (dy != dy) ? 0.f : fminf(fmaxf(dy, -64.f), 64.f);
        dx = (dx != dx) ? 0.f : fminf(fmaxf(dx, -64.f), 64.f);
        float mz = acc[18 + k2] + b_off[18 + k2];
        float mk = 1.f / (1.f + __expf(-mz));

        float py = (float)(h - 1 + k2 / 3) + dy;
        float px = (float)(w - 1 + k2 % 3) + dx;
        float fy = floorf(py), fx = floorf(px);
        float wy = py - fy,   wx = px - fx;
        int   y0 = (int)fy,   x0 = (int)fx;

        Desc d;
        auto corner = [&](int yi, int xi, float wgt, int& idx, float& wo) {
            bool v  = (yi >= 0) & (yi < Hd) & (xi >= 0) & (xi < Wd);
            int  yc = min(max(yi, 0), Hd - 1);
            int  xc = min(max(xi, 0), Wd - 1);
            idx = yc * Wd + xc;
            wo  = v ? wgt * mk : 0.f;
        };
        corner(y0,     x0,     (1.f - wy) * (1.f - wx), d.i00, d.w00);
        corner(y0,     x0 + 1, (1.f - wy) * wx,         d.i01, d.w01);
        corner(y0 + 1, x0,     wy * (1.f - wx),         d.i10, d.w10);
        corner(y0 + 1, x0 + 1, wy * wx,                 d.i11, d.w11);
        desc[dbase + k2] = d;
    }
}

// ---------------------------------------------------------------------------
// K2: deformable GEMM. One WG (256 thr = 8 waves) per 32-position M-tile.
// Each wave owns 2x2 wave-tiles (mi in {0,1}, nt in {2wv, 2wv+1}) -> 4 WMMAs
// per K-chunk. A/B LDS tiles double-buffered, 1 barrier per chunk. B staged by
// the Tensor Data Mover (wave 0 issues one tensor_load_to_lds per chunk).
// ---------------------------------------------------------------------------
__global__ __launch_bounds__(256) void k_dcn_gemm(const float* __restrict__ x,
                                                  const _Float16* __restrict__ wBT,
                                                  const Desc* __restrict__ desc,
                                                  float* __restrict__ pre) {
    __shared__ Desc sd[32][K2n];                           // 9 KB
    __shared__ __align__(64) _Float16 aT[2][32][32];       // 4 KB  (M x Kchunk)
    __shared__ __align__(64) _Float16 bT[2][256][32];      // 32 KB (N x Kchunk)

    const int tid = threadIdx.x;
    const int mt  = blockIdx.x;                            // 0..511
    const int b   = mt >> 7;
    const int hw0 = (mt & 127) << 5;                       // 32 positions

    for (int i = tid; i < 32 * K2n; i += 256) {
        int m  = i / K2n;
        int k2 = i - m * K2n;
        sd[m][k2] = desc[((size_t)b * HW + hw0 + m) * K2n + k2];
    }

    const int lane  = tid & 31;
    const int wv    = tid >> 5;                            // wave 0..7
    const int am    = tid & 31;                            // A-build row
    const int cquad = tid >> 5;                            // A-build channel quad

    const float* xb = x + (size_t)b * C1 * HW;

    // ---- stage B chunk into bT[buf] -------------------------------------
    auto stageB = [&](int chunk, int buf) {
        const int k2   = chunk % K2n;
        const int cb   = chunk / K2n;
        const int koff = k2 * 256 + cb * 32;               // element offset in a row
#if HAVE_TDM
        if (wv == 0) {
            unsigned long long ga =
                (unsigned long long)(const void*)(wBT + koff);
            unsigned lds = (unsigned)(unsigned long long)(const void*)&bT[buf][0][0];
            u32x4 g0;
            g0.x = 1u;                                     // count=1, user mode
            g0.y = lds;                                    // lds_addr
            g0.z = (unsigned)(ga & 0xFFFFFFFFull);         // global_addr[31:0]
            g0.w = (unsigned)((ga >> 32) & 0x01FFFFFFull)  // global_addr[56:32]
                 | (2u << 30);                             // type = 2 (image)
            i32x8 g1;
            g1[0] = (1 << 16);                             // data_size = 2B
            g1[1] = (int)((Ktot & 0xFFFF) << 16);          // tensor_dim0[15:0]
            g1[2] = (Ktot >> 16) | (256 << 16);            // tdim0 hi | tdim1 lo
            g1[3] = (32 << 16);                            // tile_dim0 = 32
            g1[4] = 256;                                   // tile_dim1 = 256
            g1[5] = Ktot;                                  // dim0_stride[31:0]
            g1[6] = 0;
            g1[7] = 0;
            i32x4 z4 = {0, 0, 0, 0};
#if __clang_major__ >= 23
            i32x8 z8 = {0, 0, 0, 0, 0, 0, 0, 0};
            __builtin_amdgcn_tensor_load_to_lds(g0, g1, z4, z4, z8, 0);
#else
            __builtin_amdgcn_tensor_load_to_lds(g0, g1, z4, z4, 0);
#endif
        }
#else
        const _Float16* wp = wBT + (size_t)tid * Ktot + koff;
        *(uint4*)&bT[buf][tid][0]  = *(const uint4*)(wp + 0);
        *(uint4*)&bT[buf][tid][8]  = *(const uint4*)(wp + 8);
        *(uint4*)&bT[buf][tid][16] = *(const uint4*)(wp + 16);
        *(uint4*)&bT[buf][tid][24] = *(const uint4*)(wp + 24);
        __builtin_prefetch(wp + 256, 0, 0);                // next tap slab (L2)
#endif
    };

    // ---- build A chunk (bilinear gather) into aT[buf] -------------------
    auto buildA = [&](int chunk, int buf) {
        const int k2 = chunk % K2n;
        const int cb = chunk / K2n;
        Desc dsc = sd[am][k2];
        half4 v4;
#pragma unroll
        for (int u = 0; u < 4; ++u) {
            int cl = cquad * 4 + u;
            const float* xp = xb + (size_t)(cb * 32 + cl) * HW;
            float v = dsc.w00 * xp[dsc.i00] + dsc.w01 * xp[dsc.i01]
                    + dsc.w10 * xp[dsc.i10] + dsc.w11 * xp[dsc.i11];
            v4[u] = (_Float16)v;
        }
        *(half4*)&aT[buf][am][cquad * 4] = v4;             // one 8B ds store
    };

    auto waitStage = [&]() {
#if HAVE_TDM
#  if __has_builtin(__builtin_amdgcn_s_wait_tensorcnt)
        __builtin_amdgcn_s_wait_tensorcnt(0);
#  else
        asm volatile("s_wait_tensorcnt 0x0" ::: "memory");
#  endif
#endif
        __syncthreads();
    };

    float8 acc[2][2] = {};

    // prologue: stage chunk 0 into buffer 0
    stageB(0, 0);
    buildA(0, 0);
    waitStage();

    const int abase = (lane >> 4) * 16;                    // A: lane/16 = K half
    const int khalf = (lane >> 4) * 16;                    // B: lane/16 = K half
    const int ncol  = lane & 15;                           // B: lane%16 = N column

    for (int i = 0; i < NCHUNK; ++i) {
        const int cur = i & 1;
        if (i + 1 < NCHUNK) {                              // overlap next-chunk staging
            stageB(i + 1, cur ^ 1);
            buildA(i + 1, cur ^ 1);
        }

        half16 afrag[2];
#pragma unroll
        for (int mi = 0; mi < 2; ++mi) {
            int row = mi * 16 + (lane & 15);
            half8 lo = *(const half8*)&aT[cur][row][abase];
            half8 hi = *(const half8*)&aT[cur][row][abase + 8];
            afrag[mi] = __builtin_shufflevector(lo, hi,
                0, 1, 2, 3, 4, 5, 6, 7, 8, 9, 10, 11, 12, 13, 14, 15);
        }
        half16 bfrag[2];
#pragma unroll
        for (int j = 0; j < 2; ++j) {
            int n = (wv * 2 + j) * 16 + ncol;
            bfrag[j] = *(const half16*)&bT[cur][n][khalf];
        }
#pragma unroll
        for (int mi = 0; mi < 2; ++mi)
#pragma unroll
            for (int j = 0; j < 2; ++j)
                acc[mi][j] = __builtin_amdgcn_wmma_f32_16x16x32_f16(
                    false, afrag[mi], false, bfrag[j], (short)0,
                    acc[mi][j], false, false);

        waitStage();                                       // publish buf cur^1
    }

    // ---- epilogue: D layout lane%16=N, VGPR r -> M = r + 8*(lane/16) ----
#pragma unroll
    for (int mi = 0; mi < 2; ++mi)
#pragma unroll
        for (int j = 0; j < 2; ++j) {
            int d0 = (wv * 2 + j) * 16 + ncol;
#pragma unroll
            for (int r = 0; r < 8; ++r) {
                int mm = mi * 16 + r + ((lane >> 4) << 3);
                pre[((size_t)b * HW + hw0 + mm) * C2 + d0] = acc[mi][j][r];
            }
        }
}

// ---------------------------------------------------------------------------
// K3: GroupNorm statistics: one block per (b, group), reduce 8 ch x 4096 px.
// ---------------------------------------------------------------------------
__global__ __launch_bounds__(256) void k_gn_stats(const float* __restrict__ pre,
                                                  float* __restrict__ stats) {
    const int bg = blockIdx.x;
    const int b  = bg >> 5;
    const int g  = bg & 31;
    const int tid = threadIdx.x;

    float s = 0.f, ss = 0.f;
    for (int i = tid; i < 8 * HW; i += 256) {
        int hw = i >> 3;
        int cc = i & 7;
        float v = pre[((size_t)b * HW + hw) * C2 + g * 8 + cc];
        s  += v;
        ss += v * v;
    }
    __shared__ float rs[256], rss[256];
    rs[tid] = s; rss[tid] = ss;
    __syncthreads();
    for (int st = 128; st > 0; st >>= 1) {
        if (tid < st) { rs[tid] += rs[tid + st]; rss[tid] += rss[tid + st]; }
        __syncthreads();
    }
    if (tid == 0) {
        const float inv = 1.f / (float)(8 * HW);
        float mean = rs[0] * inv;
        float var  = rss[0] * inv - mean * mean;
        stats[bg * 2]     = mean;
        stats[bg * 2 + 1] = rsqrtf(var + 1e-5f);
    }
}

// ---------------------------------------------------------------------------
// K4: normalize + affine + SiLU, transpose [b][hw][d] -> NCHW output.
// ---------------------------------------------------------------------------
__global__ __launch_bounds__(256) void k_finalize(const float* __restrict__ pre,
                                                  const float* __restrict__ stats,
                                                  const float* __restrict__ gamma,
                                                  const float* __restrict__ beta,
                                                  float* __restrict__ out) {
    int idx = blockIdx.x * 256 + threadIdx.x;              // over B*C2*HW = 2^22
    int b   = idx >> 20;
    int rem = idx & ((1 << 20) - 1);
    int d   = rem >> 12;
    int hw  = rem & (HW - 1);

    float v  = pre[((size_t)b * HW + hw) * C2 + d];
    int   g  = d >> 3;
    float mu = stats[(b * 32 + g) * 2];
    float rv = stats[(b * 32 + g) * 2 + 1];
    float o  = (v - mu) * rv * gamma[d] + beta[d];
    out[idx] = o / (1.f + __expf(-o));
}

// ---------------------------------------------------------------------------
// Launcher. Workspace map (bytes):
//   [0)            wBT   f16  C2*Ktot*2  = 1,179,648
//   [1,179,648)    desc  Desc B*HW*9*32  = 4,718,592
//   [5,898,240)    pre   f32  B*HW*C2*4  = 16,777,216
//   [22,675,456)   stats f32  128*2*4    = 1,024      (total ~22.7 MB)
// ---------------------------------------------------------------------------
extern "C" void kernel_launch(void* const* d_in, const int* in_sizes, int n_in,
                              void* d_out, int out_size, void* d_ws, size_t ws_size,
                              hipStream_t stream) {
    const float* x      = (const float*)d_in[0];
    const float* w_off  = (const float*)d_in[1];
    const float* b_off  = (const float*)d_in[2];
    const float* w_dcn  = (const float*)d_in[3];
    const float* gamma  = (const float*)d_in[4];
    const float* beta   = (const float*)d_in[5];
    float*       out    = (float*)d_out;

    char* ws = (char*)d_ws;
    _Float16* wBT   = (_Float16*)(ws + 0);
    Desc*     desc  = (Desc*)    (ws + 1179648);
    float*    pre   = (float*)   (ws + 5898240);
    float*    stats = (float*)   (ws + 22675456);

    k_pack_w  <<<(C2 * Ktot + 255) / 256, 256, 0, stream>>>(w_dcn, wBT);
    k_offconv <<<Bsz * Hd, 64, 0, stream>>>(x, w_off, b_off, desc);
    k_dcn_gemm<<<(Bsz * HW) / 32, 256, 0, stream>>>(x, wBT, desc, pre);
    k_gn_stats<<<Bsz * 32, 256, 0, stream>>>(pre, stats);
    k_finalize<<<(Bsz * C2 * HW) / 256, 256, 0, stream>>>(pre, stats, gamma, beta, out);
}
